// HiResPrecipNet_CNN_GNN_5h_1563368096257
// MI455X (gfx1250) — matrix operand order repaired and checked by reading, not compiled, not run on hardware
//
#include <hip/hip_runtime.h>

#define NLOW   60000
#define NHIGH  150000
#define ELL    480000
#define EL2H   1350000
#define EHH    1200000
#define BN_EPS 1e-5f
#define CDIV(a,b) (((a)+(b)-1)/(b))

typedef __attribute__((ext_vector_type(16))) _Float16 v16h;
typedef __attribute__((ext_vector_type(8)))  float    v8f;

// ---------- helpers ----------
static __device__ __forceinline__ unsigned encF(float f) {
  unsigned u = __float_as_uint(f);
  return (u & 0x80000000u) ? ~u : (u | 0x80000000u);
}
static __device__ __forceinline__ float decF(unsigned k) {
  unsigned u = (k & 0x80000000u) ? (k & 0x7FFFFFFFu) : ~k;
  return __uint_as_float(u);
}

__global__ void fill_f32(float* p, float v, int n) {
  int i = blockIdx.x * blockDim.x + threadIdx.x;
  if (i < n) p[i] = v;
}

// ---------- CNN encoder ----------
__global__ void cnn_conv(const float* __restrict__ x, const float* __restrict__ w,
                         const float* __restrict__ bias, float* __restrict__ y, int Nn) {
  int idx = blockIdx.x * blockDim.x + threadIdx.x;
  if (idx >= Nn * 125) return;
  int p = idx % 25, c = (idx / 25) % 5, n = idx / 125;
  int py = p / 5, px = p % 5;
  const float* xi = x + (size_t)(n * 5 + c) * 25;
  const float* wc = w + c * 9;
  float acc = bias[c];
#pragma unroll
  for (int ky = 0; ky < 3; ++ky) {
    int yy = py + ky - 1;
    if (yy < 0 || yy > 4) continue;
#pragma unroll
    for (int kx = 0; kx < 3; ++kx) {
      int xx = px + kx - 1;
      if (xx < 0 || xx > 4) continue;
      acc += xi[yy * 5 + xx] * wc[ky * 3 + kx];
    }
  }
  y[idx] = acc;
}

__global__ void cnn_stats(const float* __restrict__ x, float* __restrict__ stats, int Nn) {
  int c = blockIdx.y;
  float s = 0.f, s2 = 0.f;
  int total = Nn * 25;
  for (int i = blockIdx.x * blockDim.x + threadIdx.x; i < total; i += gridDim.x * blockDim.x) {
    int n = i / 25, p = i % 25;
    float v = x[(size_t)(n * 5 + c) * 25 + p];
    s += v; s2 += v * v;
  }
  __shared__ float sh[256], sh2[256];
  sh[threadIdx.x] = s; sh2[threadIdx.x] = s2;
  __syncthreads();
  for (int d = 128; d > 0; d >>= 1) {
    if ((int)threadIdx.x < d) { sh[threadIdx.x] += sh[threadIdx.x + d]; sh2[threadIdx.x] += sh2[threadIdx.x + d]; }
    __syncthreads();
  }
  if (threadIdx.x == 0) { atomicAdd(&stats[c], sh[0]); atomicAdd(&stats[5 + c], sh2[0]); }
}

__global__ void cnn_bn_relu(float* x, const float* __restrict__ stats,
                            const float* __restrict__ g, const float* __restrict__ b, int Nn) {
  int idx = blockIdx.x * blockDim.x + threadIdx.x;
  if (idx >= Nn * 125) return;
  int c = (idx / 25) % 5;
  float cntf = (float)Nn * 25.f;
  float m = stats[c] / cntf;
  float v = stats[5 + c] / cntf - m * m;
  float t = (x[idx] - m) * rsqrtf(v + BN_EPS) * g[c] + b[c];
  x[idx] = fmaxf(t, 0.f);
}

__global__ void maxpool_flat(const float* __restrict__ x, float* __restrict__ y, int Nn) {
  int idx = blockIdx.x * blockDim.x + threadIdx.x;
  if (idx >= Nn * 45) return;
  int f = idx % 45, n = idx / 45;
  int c = f / 9, r = (f % 9) / 3, cc = f % 3;
  const float* xi = x + (size_t)(n * 5 + c) * 25;
  float m = -3.4e38f;
  int r0 = 2 * r - 1, c0 = 2 * cc - 1;
#pragma unroll
  for (int dy = 0; dy < 2; ++dy)
#pragma unroll
    for (int dx = 0; dx < 2; ++dx) {
      int yy = r0 + dy, xx = c0 + dx;
      if (yy >= 0 && yy < 5 && xx >= 0 && xx < 5) m = fmaxf(m, xi[yy * 5 + xx]);
    }
  y[idx] = m;
}

// ---------- WMMA GEMM:  Y[M,N] = (relu?)(X[M,K] @ W[K,N] + b[N]) ----------
// Wave owns a 16x32 output tile (two 16x16 WMMA tiles sharing the A fragment).
// OOB rows/cols handled by address clamping (results discarded at store);
// K-tail handled branchlessly with select-to-zero. Hot loop: loads -> wait -> 2x wmma.
__global__ void gemm_wmma(const float* __restrict__ X, const float* __restrict__ W,
                          const float* __restrict__ b, float* __restrict__ Y,
                          int M, int K, int N, int relu) {
  int wave = threadIdx.x >> 5;
  int lane = threadIdx.x & 31;
  int m0 = (blockIdx.x * 4 + wave) * 16;
  int n0 = blockIdx.y * 32;
  if (m0 >= M) return;
  int r    = lane & 15;
  int half = lane >> 4;
  int ahi  = half ? 16 : 0;  // A: lanes 0-15 -> K 0..15, lanes 16-31 -> K 16..31
  int bhi  = half ? 16 : 0;  // B: lanes 0-15 -> K 0..15, lanes 16-31 -> K 16..31
  int row  = m0 + r;
  int rowS = row < M ? row : M - 1;
  int col0 = n0 + r;
  int col1 = n0 + 16 + r;
  int col0S = col0 < N ? col0 : N - 1;
  int col1S = col1 < N ? col1 : N - 1;
  const float* Xr = X + (size_t)rowS * K;
  v8f acc0 = {}, acc1 = {};

  int kfull = K & ~31;
  int kb = 0;
  for (; kb < kfull; kb += 32) {        // branchless steady state
    v16h a, b0, b1;
#pragma unroll
    for (int i = 0; i < 8; ++i) {
      a[i]     = (_Float16)Xr[kb + ahi + i];
      a[i + 8] = (_Float16)Xr[kb + 8 + ahi + i];
    }
#pragma unroll
    for (int i = 0; i < 16; ++i) {
      size_t kr = (size_t)(kb + bhi + i) * N;
      b0[i] = (_Float16)W[kr + col0S];
      b1[i] = (_Float16)W[kr + col1S];
    }
    acc0 = __builtin_amdgcn_wmma_f32_16x16x32_f16(false, a, false, b0, (short)0, acc0, false, false);
    acc1 = __builtin_amdgcn_wmma_f32_16x16x32_f16(false, a, false, b1, (short)0, acc1, false, false);
  }
  if (kb < K) {                          // branchless tail (K = 45, 65, ...)
    v16h a, b0, b1;
#pragma unroll
    for (int i = 0; i < 8; ++i) {
      int k0 = kb + ahi + i;
      int k1 = kb + 8 + ahi + i;
      float v0 = Xr[k0 < K ? k0 : K - 1];
      float v1 = Xr[k1 < K ? k1 : K - 1];
      a[i]     = (_Float16)(k0 < K ? v0 : 0.f);
      a[i + 8] = (_Float16)(k1 < K ? v1 : 0.f);
    }
#pragma unroll
    for (int i = 0; i < 16; ++i) {
      int k = kb + bhi + i;
      size_t kr = (size_t)(k < K ? k : K - 1) * N;
      float w0 = W[kr + col0S];
      float w1 = W[kr + col1S];
      b0[i] = (_Float16)(k < K ? w0 : 0.f);
      b1[i] = (_Float16)(k < K ? w1 : 0.f);
    }
    acc0 = __builtin_amdgcn_wmma_f32_16x16x32_f16(false, a, false, b0, (short)0, acc0, false, false);
    acc1 = __builtin_amdgcn_wmma_f32_16x16x32_f16(false, a, false, b1, (short)0, acc1, false, false);
  }

  int mbase = m0 + (half ? 8 : 0);
  if (col0 < N) {
    float bias = b[col0];
#pragma unroll
    for (int j = 0; j < 8; ++j) {
      int m = mbase + j;
      if (m < M) {
        float v = acc0[j] + bias;
        if (relu) v = fmaxf(v, 0.f);
        Y[(size_t)m * N + col0] = v;
      }
    }
  }
  if (col1 < N) {
    float bias = b[col1];
#pragma unroll
    for (int j = 0; j < 8; ++j) {
      int m = mbase + j;
      if (m < M) {
        float v = acc1[j] + bias;
        if (relu) v = fmaxf(v, 0.f);
        Y[(size_t)m * N + col1] = v;
      }
    }
  }
}

// K==1 linear (downscaler target transform)
__global__ void outer1(const float* __restrict__ xh, const float* __restrict__ w,
                       const float* __restrict__ b, float* __restrict__ y, int Nn, int C) {
  int idx = blockIdx.x * blockDim.x + threadIdx.x;
  if (idx >= Nn * C) return;
  int c = idx % C, n = idx / C;
  y[idx] = xh[n] * w[c] + b[c];
}

// ---------- GATv2 edge pipeline ----------
// wave-per-(edge,head): lanes cover the C dim (coalesced row gathers), shfl reduce
__global__ void gat_logits(const float* __restrict__ xl, const float* __restrict__ xr,
                           const int* __restrict__ src, const int* __restrict__ dst,
                           const float* __restrict__ att, float* __restrict__ elog,
                           int E, int Ndst, int selfLoops, int H, int C) {
  int Etot = E + (selfLoops ? Ndst : 0);
  int gt = blockIdx.x * blockDim.x + threadIdx.x;
  int w = gt >> 5, lane = gt & 31;
  if (w >= Etot * H) return;
  int e = w / H, h = w - e * H;
  int s = (e < E) ? src[e] : (e - E);
  int d = (e < E) ? dst[e] : (e - E);
  const float* pl = xl + ((size_t)s * H + h) * C;
  const float* pr = xr + ((size_t)d * H + h) * C;
  const float* pa = att + (size_t)h * C;
  float acc = 0.f;
  for (int c = lane; c < C; c += 32) {
    float v = pl[c] + pr[c];
    v = v > 0.f ? v : 0.2f * v;          // leaky_relu(0.2)
    acc += v * pa[c];
  }
#pragma unroll
  for (int mask = 16; mask > 0; mask >>= 1) acc += __shfl_xor(acc, mask, 32);
  if (lane == 0) elog[w] = acc;
}

__global__ void gat_max(const float* __restrict__ elog, const int* __restrict__ dst,
                        unsigned* __restrict__ emax, int E, int Ndst, int selfLoops, int H) {
  int Etot = E + (selfLoops ? Ndst : 0);
  int t = blockIdx.x * blockDim.x + threadIdx.x;
  if (t >= Etot * H) return;
  int e = t / H, h = t - e * H;
  int d = (e < E) ? dst[e] : (e - E);
  atomicMax(&emax[(size_t)d * H + h], encF(elog[t]));
}

__global__ void gat_expdenom(float* __restrict__ elog, const int* __restrict__ dst,
                             const unsigned* __restrict__ emax, float* __restrict__ denom,
                             float* __restrict__ cnt, int E, int Ndst, int selfLoops, int H) {
  int Etot = E + (selfLoops ? Ndst : 0);
  int t = blockIdx.x * blockDim.x + threadIdx.x;
  if (t >= Etot * H) return;
  int e = t / H, h = t - e * H;
  int d = (e < E) ? dst[e] : (e - E);
  float ex = expf(elog[t] - decF(emax[(size_t)d * H + h]));
  elog[t] = ex;
  atomicAdd(&denom[(size_t)d * H + h], ex);
  if (h == 0) atomicAdd(&cnt[d], 1.0f);
}

// wave-per-(edge,head): lanes do coalesced atomic adds across the feature dim
__global__ void gat_agg(const float* __restrict__ ex, const float* __restrict__ denom,
                        const float* __restrict__ xl, const int* __restrict__ src,
                        const int* __restrict__ dst, float* __restrict__ sbuf,
                        int E, int Ndst, int selfLoops, int H, int C) {
  int Etot = E + (selfLoops ? Ndst : 0);
  int gt = blockIdx.x * blockDim.x + threadIdx.x;
  int w = gt >> 5, lane = gt & 31;
  if (w >= Etot * H) return;
  int e = w / H, h = w - e * H;
  int s = (e < E) ? src[e] : (e - E);
  int d = (e < E) ? dst[e] : (e - E);
  float alpha = ex[w] / denom[(size_t)d * H + h];
  const float* xs = xl + ((size_t)s * H + h) * C;
  float* sd = sbuf + ((size_t)d * H + h) * C;
  for (int c = lane; c < C; c += 32) atomicAdd(&sd[c], alpha * xs[c]);
}

__global__ void gat_final(const float* __restrict__ sbuf, const float* __restrict__ cnt,
                          const float* __restrict__ bias, float* __restrict__ out,
                          int Ndst, int HC, int relu) {
  int idx = blockIdx.x * blockDim.x + threadIdx.x;
  if (idx >= Ndst * HC) return;
  int f = idx % HC, d = idx / HC;
  float v = sbuf[idx] / fmaxf(cnt[d], 1.f) + bias[f];
  out[idx] = relu ? fmaxf(v, 0.f) : v;
}

// ---------- graph BatchNorm over node dim ----------
__global__ void bn_stats(const float* __restrict__ x, float* __restrict__ stats, int Nn, int F) {
  int f = threadIdx.x;                    // blockDim.x == F
  float s = 0.f, s2 = 0.f;
  for (int n = blockIdx.x; n < Nn; n += gridDim.x) {
    float v = x[(size_t)n * F + f];
    s += v; s2 += v * v;
  }
  atomicAdd(&stats[f], s);
  atomicAdd(&stats[F + f], s2);
}

__global__ void bn_apply(float* __restrict__ x, const float* __restrict__ stats,
                         const float* __restrict__ g, const float* __restrict__ b,
                         int Nn, int F, int relu) {
  int idx = blockIdx.x * blockDim.x + threadIdx.x;
  if (idx >= Nn * F) return;
  int f = idx % F;
  float m = stats[f] / (float)Nn;
  float v = stats[F + f] / (float)Nn - m * m;
  float t = (x[idx] - m) * rsqrtf(v + BN_EPS) * g[f] + b[f];
  x[idx] = relu ? fmaxf(t, 0.f) : t;
}

__global__ void concat_z(const float* __restrict__ z, const float* __restrict__ ds,
                         float* __restrict__ h, int Nn) {
  int idx = blockIdx.x * blockDim.x + threadIdx.x;
  if (idx >= Nn * 65) return;
  int f = idx % 65, n = idx / 65;
  h[idx] = (f == 0) ? z[n] : ds[(size_t)n * 64 + (f - 1)];
}

__global__ void rowdot(const float* __restrict__ x, const float* __restrict__ w,
                       const float* __restrict__ b, float* __restrict__ y, int Nn, int K) {
  int n = blockIdx.x * blockDim.x + threadIdx.x;
  if (n >= Nn) return;
  float acc = b[0];
  for (int k = 0; k < K; ++k) acc += x[(size_t)n * K + k] * w[k];
  y[n] = acc;
}

// ---------- host orchestration ----------
extern "C" void kernel_launch(void* const* d_in, const int* in_sizes, int n_in,
                              void* d_out, int out_size, void* d_ws, size_t ws_size,
                              hipStream_t stream) {
  (void)in_sizes; (void)n_in; (void)out_size; (void)ws_size;
  const float* x_low   = (const float*)d_in[0];
  const float* x_high  = (const float*)d_in[1];
  const float* z_std   = (const float*)d_in[2];
  const int* e_ll_src  = (const int*)d_in[3];
  const int* e_ll_dst  = (const int*)d_in[4];
  const int* e_l2h_src = (const int*)d_in[5];
  const int* e_l2h_dst = (const int*)d_in[6];
  const int* e_hh_src  = (const int*)d_in[7];
  const int* e_hh_dst  = (const int*)d_in[8];
  const float* conv_w = (const float*)d_in[9];
  const float* conv_b = (const float*)d_in[10];
  const float* bn2d_g = (const float*)d_in[11];
  const float* bn2d_b = (const float*)d_in[12];
  const float* pl_Wl = (const float*)d_in[13];
  const float* pl_bl = (const float*)d_in[14];
  const float* pl_Wr = (const float*)d_in[15];
  const float* pl_br = (const float*)d_in[16];
  const float* pl_att = (const float*)d_in[17];
  const float* pl_bias = (const float*)d_in[18];
  const float* ds_Wl = (const float*)d_in[19];
  const float* ds_bl = (const float*)d_in[20];
  const float* ds_Wr = (const float*)d_in[21];
  const float* ds_br = (const float*)d_in[22];
  const float* ds_att = (const float*)d_in[23];
  const float* ds_bias = (const float*)d_in[24];
  const float* bn_g0 = (const float*)d_in[25];
  const float* bn_b0 = (const float*)d_in[26];
  const float* bn_g = (const float*)d_in[27];
  const float* bn_b = (const float*)d_in[28];
  const float* p1_Wl = (const float*)d_in[29];
  const float* p1_bl = (const float*)d_in[30];
  const float* p1_Wr = (const float*)d_in[31];
  const float* p1_br = (const float*)d_in[32];
  const float* p1_att = (const float*)d_in[33];
  const float* p1_bias = (const float*)d_in[34];
  const float* pm_Wl = (const float*)d_in[35];
  const float* pm_bl = (const float*)d_in[36];
  const float* pm_Wr = (const float*)d_in[37];
  const float* pm_br = (const float*)d_in[38];
  const float* pm_att = (const float*)d_in[39];
  const float* pm_bias = (const float*)d_in[40];
  const float* p5_Wl = (const float*)d_in[41];
  const float* p5_bl = (const float*)d_in[42];
  const float* p5_Wr = (const float*)d_in[43];
  const float* p5_br = (const float*)d_in[44];
  const float* p5_att = (const float*)d_in[45];
  const float* p5_bias = (const float*)d_in[46];
  const float* pr_W1 = (const float*)d_in[47];
  const float* pr_b1 = (const float*)d_in[48];
  const float* pr_W2 = (const float*)d_in[49];
  const float* pr_b2 = (const float*)d_in[50];
  const float* pr_W3 = (const float*)d_in[51];
  const float* pr_b3 = (const float*)d_in[52];

  // workspace layout
  char* wsp = (char*)d_ws;
  size_t off = 0;
  auto walloc = [&](size_t bytes) -> void* {
    void* p = wsp + off;
    off += (bytes + 255) & ~(size_t)255;
    return p;
  };
  float* cnnA = (float*)walloc((size_t)NLOW * 125 * 4);
  float* cnnB = (float*)walloc((size_t)NLOW * 125 * 4);
  float* lowA = (float*)walloc((size_t)NLOW * 45 * 4);
  float* lowB = (float*)walloc((size_t)NLOW * 45 * 4);
  float* xlB  = (float*)walloc((size_t)NHIGH * 128 * 4);
  float* xrB  = (float*)walloc((size_t)NHIGH * 128 * 4);
  float* hA   = (float*)walloc((size_t)NHIGH * 128 * 4);
  float* hB   = (float*)walloc((size_t)NHIGH * 128 * 4);
  float* sAgg = (float*)walloc((size_t)NHIGH * 128 * 4);
  float* elog = (float*)walloc((size_t)(EL2H * 2) * 4);  // >= max Etot*H
  unsigned* emax = (unsigned*)walloc((size_t)NHIGH * 2 * 4);
  float* denom = (float*)walloc((size_t)NHIGH * 2 * 4);
  float* cnt   = (float*)walloc((size_t)NHIGH * 4);
  float* stats = (float*)walloc(1024 * 4);

  auto zero = [&](float* p, int n) {
    fill_f32<<<CDIV(n, 256), 256, 0, stream>>>(p, 0.f, n);
  };
  auto gemm = [&](const float* X, const float* W, const float* b, float* Y,
                  int M, int K, int N, int relu) {
    dim3 g(CDIV(M, 64), CDIV(N, 32));
    gemm_wmma<<<g, 128, 0, stream>>>(X, W, b, Y, M, K, N, relu);
  };
  auto runGAT = [&](const float* xl, const float* xr, const int* src, const int* dst,
                    int E, int Ndst, int selfLoops, const float* att, int H, int C,
                    const float* bias, float* out, int relu) {
    int Etot = E + (selfLoops ? Ndst : 0);
    int HC = H * C;
    zero((float*)emax, Ndst * H);
    zero(denom, Ndst * H);
    zero(cnt, Ndst);
    zero(sAgg, Ndst * HC);
    int nt = Etot * H;                       // (edge, head) work items
    int nwBlocks = CDIV(nt, 8);              // 8 waves / 256-thread block
    gat_logits<<<nwBlocks, 256, 0, stream>>>(xl, xr, src, dst, att, elog, E, Ndst, selfLoops, H, C);
    gat_max<<<CDIV(nt, 256), 256, 0, stream>>>(elog, dst, emax, E, Ndst, selfLoops, H);
    gat_expdenom<<<CDIV(nt, 256), 256, 0, stream>>>(elog, dst, emax, denom, cnt, E, Ndst, selfLoops, H);
    gat_agg<<<nwBlocks, 256, 0, stream>>>(elog, denom, xl, src, dst, sAgg, E, Ndst, selfLoops, H, C);
    gat_final<<<CDIV(Ndst * HC, 256), 256, 0, stream>>>(sAgg, cnt, bias, out, Ndst, HC, relu);
  };
  auto runBN = [&](float* x, const float* g, const float* b, int Nn, int F, int relu) {
    zero(stats, 2 * F);
    bn_stats<<<dim3(256), dim3(F), 0, stream>>>(x, stats, Nn, F);
    bn_apply<<<CDIV(Nn * F, 256), 256, 0, stream>>>(x, stats, g, b, Nn, F, relu);
  };

  // ---- CNN encoder ----
  const float* cin = x_low;
  for (int i = 0; i < 3; ++i) {
    float* cout = (i == 1) ? cnnB : cnnA;
    cnn_conv<<<CDIV(NLOW * 125, 256), 256, 0, stream>>>(cin, conv_w + i * 45, conv_b + i * 5, cout, NLOW);
    zero(stats, 10);
    cnn_stats<<<dim3(256, 5), 256, 0, stream>>>(cout, stats, NLOW);
    cnn_bn_relu<<<CDIV(NLOW * 125, 256), 256, 0, stream>>>(cout, stats, bn2d_g + i * 5, bn2d_b + i * 5, NLOW);
    cin = cout;
  }
  maxpool_flat<<<CDIV(NLOW * 45, 256), 256, 0, stream>>>(cnnA, lowA, NLOW);

  // ---- low-res processor: 3x (GATv2 h=1,c=45 + ReLU), no self loops ----
  float* xcur = lowA; float* xnxt = lowB;
  for (int i = 0; i < 3; ++i) {
    gemm(xcur, pl_Wl + i * 2025, pl_bl + i * 45, xlB, NLOW, 45, 45, 0);
    gemm(xcur, pl_Wr + i * 2025, pl_br + i * 45, xrB, NLOW, 45, 45, 0);
    runGAT(xlB, xrB, e_ll_src, e_ll_dst, ELL, NLOW, 0, pl_att + i * 45, 1, 45,
           pl_bias + i * 45, xnxt, 1);
    float* t = xcur; xcur = xnxt; xnxt = t;
  }

  // ---- bipartite downscaler low->high ----
  gemm(xcur, ds_Wl, ds_bl, xlB, NLOW, 45, 64, 0);
  outer1<<<CDIV(NHIGH * 64, 256), 256, 0, stream>>>(x_high, ds_Wr, ds_br, xrB, NHIGH, 64);
  runGAT(xlB, xrB, e_l2h_src, e_l2h_dst, EL2H, NHIGH, 0, ds_att, 1, 64, ds_bias, hA, 0);

  // ---- concat + BN ----
  concat_z<<<CDIV(NHIGH * 65, 256), 256, 0, stream>>>(z_std, hA, hB, NHIGH);
  runBN(hB, bn_g0, bn_b0, NHIGH, 65, 0);

  // ---- high-res processor ----
  gemm(hB, p1_Wl, p1_bl, xlB, NHIGH, 65, 128, 0);
  gemm(hB, p1_Wr, p1_br, xrB, NHIGH, 65, 128, 0);
  runGAT(xlB, xrB, e_hh_src, e_hh_dst, EHH, NHIGH, 1, p1_att, 2, 64, p1_bias, hA, 0);
  runBN(hA, bn_g, bn_b, NHIGH, 128, 1);

  float* hcur = hA; float* hnxt = hB;
  for (int i = 0; i < 3; ++i) {
    gemm(hcur, pm_Wl + i * 16384, pm_bl + i * 128, xlB, NHIGH, 128, 128, 0);
    gemm(hcur, pm_Wr + i * 16384, pm_br + i * 128, xrB, NHIGH, 128, 128, 0);
    runGAT(xlB, xrB, e_hh_src, e_hh_dst, EHH, NHIGH, 1, pm_att + i * 128, 2, 64,
           pm_bias + i * 128, hnxt, 0);
    runBN(hnxt, bn_g + (i + 1) * 128, bn_b + (i + 1) * 128, NHIGH, 128, 1);
    float* t = hcur; hcur = hnxt; hnxt = t;
  }

  gemm(hcur, p5_Wl, p5_bl, xlB, NHIGH, 128, 64, 0);
  gemm(hcur, p5_Wr, p5_br, xrB, NHIGH, 128, 64, 0);
  runGAT(xlB, xrB, e_hh_src, e_hh_dst, EHH, NHIGH, 1, p5_att, 1, 64, p5_bias, hnxt, 1);

  // ---- predictor MLP ----
  gemm(hnxt, pr_W1, pr_b1, xlB, NHIGH, 64, 64, 1);
  gemm(xlB, pr_W2, pr_b2, xrB, NHIGH, 64, 32, 1);
  rowdot<<<CDIV(NHIGH, 256), 256, 0, stream>>>(xrB, pr_W3, pr_b3, (float*)d_out, NHIGH, 32);
}
